// InteractionLayer_65730179498522
// MI455X (gfx1250) — compile-verified, matrix operand order checked by reading, not run on hardware
//
#include <hip/hip_runtime.h>
#include <math.h>

// ---------------- problem constants ----------------
constexpr int N_NODES = 10000;
constexpr int E_EDGES = 320000;
constexpr int M_HEADS = 32;     // M

typedef __attribute__((ext_vector_type(16))) _Float16 v16h;
typedef __attribute__((ext_vector_type(8)))  _Float16 v8h;
typedef __attribute__((ext_vector_type(8)))  float    v8f;
typedef __attribute__((ext_vector_type(4)))  float    f4v;

// ---------------- WMMA helpers (wave32, 16x16x32 f16 -> f32) ----------------
// A tile lives in LDS row-major f16 (16 rows, stride S halves).
// ISA A layout: lane l (l<16 -> M=l, K in {k0..k0+7, k0+16..k0+23};
//               l>=16 -> M=l-16, K in {k0+8..k0+15, k0+24..k0+31}).
__device__ __forceinline__ v16h load_A_lds(const _Float16* t, int S, int k0) {
  int l  = threadIdx.x & 31;
  int r  = l & 15;
  int hi = l >> 4;
  const _Float16* p = t + r * S + k0 + hi * 8;
  v8h lo  = *(const v8h*)p;
  v8h hiv = *(const v8h*)(p + 16);
  v16h a;
#pragma unroll
  for (int i = 0; i < 8; ++i) { a[i] = lo[i]; a[i + 8] = hiv[i]; }
  return a;
}

// B operand pre-packed so each lane reads 16 contiguous halves (32B).
__device__ __forceinline__ v16h load_B_packed(const _Float16* pw, int ksteps,
                                              int nslab, int kstep) {
  int l = threadIdx.x & 31;
  return *(const v16h*)(pw + (((size_t)nslab * ksteps + kstep) * 32 + l) * 16);
}

// C = A(16xK) * B(KxNslab16) accumulated in f32; K multiple of 32.
__device__ __forceinline__ v8f gemm_tile(const _Float16* atile, int S, int K,
                                         const _Float16* pB, int nslab) {
  v8f c = {0.f, 0.f, 0.f, 0.f, 0.f, 0.f, 0.f, 0.f};
  int ks = K >> 5;
  for (int t = 0; t < ks; ++t) {
    v16h a = load_A_lds(atile, S, t * 32);
    v16h b = load_B_packed(pB, ks, nslab, t);
    c = __builtin_amdgcn_wmma_f32_16x16x32_f16(false, a, false, b, (short)0, c,
                                               false, false);
  }
  return c;
}

__device__ __forceinline__ float silu(float x) { return x / (1.f + __expf(-x)); }

// streaming (single-use) access helpers -> TH=NT cache policy, protects L2
__device__ __forceinline__ float ldnt(const float* p) { return __builtin_nontemporal_load(p); }
__device__ __forceinline__ f4v  ldnt4(const float* p) { return __builtin_nontemporal_load((const f4v*)p); }
__device__ __forceinline__ void stnt(float* p, float v) { __builtin_nontemporal_store(v, p); }

// ---------------- weight repack: row-major f32 W[K x Nout] -> WMMA-B f16 ----
__global__ void pack_weights_k(const float* __restrict__ W, _Float16* __restrict__ out,
                               int K, int Nout) {
  int idx = blockIdx.x * blockDim.x + threadIdx.x;
  int total = K * Nout;
  if (idx >= total) return;
  int ksteps = K >> 5;
  int tile = idx >> 9;       // 512 halves per (nslab,kstep) tile
  int rem  = idx & 511;
  int lane = rem >> 4;
  int i    = rem & 15;
  int nslab = tile / ksteps;
  int kstep = tile - nslab * ksteps;
  int n  = nslab * 16 + (lane & 15);
  int hi = lane >> 4;
  int k  = kstep * 32 + hi * 8 + (i < 8 ? i : (i - 8) + 16);
  out[idx] = (_Float16)W[(size_t)k * Nout + n];
}

// ---------------- zero accumulators ----------------
__global__ void zero_k(float* __restrict__ p, long n) {
  long i = (long)blockIdx.x * blockDim.x + threadIdx.x;
  if (i < n) p[i] = 0.f;
}

// ---------------- per-node Q: Qn[N x 512] = node_inv[N x 64] @ Wq ----------
// Qn stays RT-cached: 20MB table, gathered by every edge in phase 1 (L2-resident).
__global__ void qn_k(const float* __restrict__ nodeinv, const _Float16* __restrict__ pWq,
                     float* __restrict__ Qn) {
  __shared__ _Float16 A[4][16 * 64];
  int wv = threadIdx.x >> 5, l = threadIdx.x & 31;
  int tile = blockIdx.x * 4 + wv;
  if (tile > 624) tile = 624;  // clamp (redundant recompute, no divergence at barrier)
  int r0 = tile * 16;
  _Float16* At = A[wv];
  for (int idx = l; idx < 16 * 16; idx += 32) {   // 16 rows x 16 f4 chunks
    int r = idx >> 4, c4 = (idx & 15) * 4;
    f4v v = ldnt4(nodeinv + (size_t)(r0 + r) * 64 + c4);
#pragma unroll
    for (int j = 0; j < 4; ++j) At[r * 64 + c4 + j] = (_Float16)v[j];
  }
  __syncthreads();
  for (int ns = 0; ns < 32; ++ns) {
    v8f c = gemm_tile(At, 64, 64, pWq, ns);
    int col = ns * 16 + (l & 15);
#pragma unroll
    for (int r = 0; r < 8; ++r) {
      int row = r + ((l >> 4) << 3);
      Qn[(size_t)(r0 + row) * 512 + col] = c[r];
    }
  }
}

// ---------------- phase 1: env MLP + gates + K + Q.K -> w, denom -----------
#define W1 4
__global__ void phase1_k(const float* __restrict__ scalar_latent,
                         const float* __restrict__ equiv,
                         const int* __restrict__ edge_center,
                         const _Float16* __restrict__ pWenv1, const float* __restrict__ benv1,
                         const _Float16* __restrict__ pWenv2, const float* __restrict__ benv2,
                         const _Float16* __restrict__ pWk,
                         const float* __restrict__ Qn,
                         float* __restrict__ gates_ws, float* __restrict__ w_ws,
                         float* __restrict__ denom) {
  __shared__ _Float16 featS[W1][16 * 192];
  __shared__ _Float16 Hs[W1][16 * 128];
  int wv = threadIdx.x >> 5, l = threadIdx.x & 31;
  long tile = (long)blockIdx.x * W1 + wv;   // 20000 tiles exactly
  long e0 = tile * 16;
  _Float16* F = featS[wv];
  _Float16* H = Hs[wv];

  // Stage A tile: [scalar(128) | x0(32) | |xv|(32)] as f16 (streaming NT reads).
  for (int idx = l; idx < 16 * 32; idx += 32) {  // 16 rows x 32 f4 chunks
    int r = idx >> 5, c4 = (idx & 31) * 4;
    f4v v = ldnt4(scalar_latent + (e0 + r) * 128 + c4);
#pragma unroll
    for (int j = 0; j < 4; ++j) F[r * 192 + c4 + j] = (_Float16)v[j];
  }
  for (int idx = l; idx < 16 * 32; idx += 32) {
    int r = idx >> 5, m = idx & 31;
    f4v q = ldnt4(equiv + ((e0 + r) * 32 + m) * 4);
    F[r * 192 + 128 + m] = (_Float16)q[0];
    F[r * 192 + 160 + m] = (_Float16)sqrtf(q[1] * q[1] + q[2] * q[2] + q[3] * q[3]);
  }
  __syncthreads();

  // GEMM1: 192 -> 128, +bias, silu -> H (f16 LDS)
  for (int ns = 0; ns < 8; ++ns) {
    v8f c = gemm_tile(F, 192, 192, pWenv1, ns);
    int col = ns * 16 + (l & 15);
    float b = benv1[col];
#pragma unroll
    for (int r = 0; r < 8; ++r) {
      int row = r + ((l >> 4) << 3);
      H[row * 128 + col] = (_Float16)silu(c[r] + b);
    }
  }
  __syncthreads();

  // GEMM2: gates = H @ Wenv2 + b (128 -> 64). Default RT stores: gates+w
  // (123MB) are re-read by the immediately following scatter pass -> keep in L2.
  for (int ns = 0; ns < 4; ++ns) {
    v8f c = gemm_tile(H, 128, 128, pWenv2, ns);
    int col = ns * 16 + (l & 15);
    float b = benv2[col];
#pragma unroll
    for (int r = 0; r < 8; ++r) {
      int row = r + ((l >> 4) << 3);
      gates_ws[(e0 + row) * 64 + col] = c[r] + b;
    }
  }

  // K per m-slab (cols m*16..m*16+15 of Wk are exactly head m), dot with Qn.
  // Clip(+-5) makes the segment-max subtraction redundant -> w = exp(logit).
  for (int m = 0; m < M_HEADS; ++m) {
    v8f c = gemm_tile(F, 192, 128, pWk, m);  // A = scalar part (first 128 cols)
    int d = l & 15;
#pragma unroll
    for (int r = 0; r < 8; ++r) {
      int row = r + ((l >> 4) << 3);
      int ctr = edge_center[e0 + row];
      float p = c[r] * Qn[(size_t)ctr * 512 + m * 16 + d];   // L2-resident gather
      p += __shfl_xor(p, 1, 32);
      p += __shfl_xor(p, 2, 32);
      p += __shfl_xor(p, 4, 32);
      p += __shfl_xor(p, 8, 32);   // sum over the 16 d-lanes of this half
      if (d == 0) {
        float lg = fminf(5.f, fmaxf(-5.f, p * 0.25f));  // 1/sqrt(16)
        float w = __expf(lg);
        w_ws[(e0 + row) * 32 + m] = w;
        atomicAdd(&denom[ctr * 32 + m], w);
      }
    }
  }
}

// ---------------- scatter: attn-weighted env into node accumulators --------
__global__ void scatter_k(const float* __restrict__ equiv, const float* __restrict__ gates,
                          const float* __restrict__ w_ws, const float* __restrict__ denom,
                          const int* __restrict__ edge_center,
                          float* __restrict__ n0acc, float* __restrict__ nvacc) {
  long idx = (long)blockIdx.x * blockDim.x + threadIdx.x;  // E*M
  if (idx >= (long)E_EDGES * M_HEADS) return;
  long e = idx >> 5;
  int m = (int)(idx & 31);
  int ctr = edge_center[e];
  float attn = w_ws[idx] / denom[ctr * 32 + m];
  f4v q = ldnt4(equiv + idx * 4);          // streaming
  float g0 = gates[e * 64 + m];            // L2 hits from phase 1
  float g1 = gates[e * 64 + 32 + m];
  atomicAdd(&n0acc[ctr * 32 + m], q[0] * g0 * attn);
  float gv = g1 * attn;
  atomicAdd(&nvacc[(ctr * 32 + m) * 3 + 0], q[1] * gv);
  atomicAdd(&nvacc[(ctr * 32 + m) * 3 + 1], q[2] * gv);
  atomicAdd(&nvacc[(ctr * 32 + m) * 3 + 2], q[3] * gv);
}

// ---------------- per-node RMS norms (in place) ----------------------------
__global__ void nodenorm_k(float* __restrict__ n0, float* __restrict__ nv) {
  int n = blockIdx.x * blockDim.x + threadIdx.x;
  if (n >= N_NODES) return;
  float ss = 0.f;
  for (int m = 0; m < 32; ++m) { float v = n0[n * 32 + m]; ss += v * v; }
  float i0 = rsqrtf(ss / 32.f + 1e-5f);
  for (int m = 0; m < 32; ++m) n0[n * 32 + m] *= i0;
  float sv = 0.f;
  for (int i = 0; i < 96; ++i) { float v = nv[n * 96 + i]; sv += v * v; }
  float iv = rsqrtf(sv / 96.f + 1e-5f);
  for (int i = 0; i < 96; ++i) nv[n * 96 + i] *= iv;
}

// ---------------- phase 2: pair features, Wp1/Wp2 MLP, blended outputs -----
#define W2 2
__global__ void phase2_k(const float* __restrict__ scalar_latent,
                         const float* __restrict__ equiv,
                         const float* __restrict__ cond,
                         const int* __restrict__ edge_center,
                         const float* __restrict__ n0, const float* __restrict__ nv,
                         const _Float16* __restrict__ pWp1, const float* __restrict__ bp1,
                         const _Float16* __restrict__ pWp2, const float* __restrict__ bp2,
                         const float* __restrict__ coeffp,
                         float* __restrict__ scalar_out, float* __restrict__ equiv_out) {
  __shared__ _Float16 featS[W2][16 * 96];
  __shared__ _Float16 Hs[W2][16 * 256];
  __shared__ float SnS[W2][16 * 64];    // s0n | s1n
  __shared__ float VnS[W2][16 * 192];   // v0n(96) | v1n(96)
  int wv = threadIdx.x >> 5, l = threadIdx.x & 31;
  long tile = (long)blockIdx.x * W2 + wv;  // 20000 tiles exactly
  long e0 = tile * 16;
  _Float16* F = featS[wv];
  _Float16* H = Hs[wv];
  float* Sn = SnS[wv];
  float* Vn = VnS[wv];

  // Feature build: 2 lanes per edge row split even/odd m; partial RMS sums
  // combined with a single xor-16 butterfly (lanes l and l^16 share a row).
  {
    int row = l & 15;
    int par = l >> 4;
    long e = e0 + row;
    int ctr = edge_center[e];
    float ss0 = 0.f, ss1 = 0.f, sv0 = 0.f, sv1 = 0.f;
    for (int m = par; m < 32; m += 2) {
      f4v q = ldnt4(equiv + (e * 32 + m) * 4);
      float x0 = q[0], xa = q[1], xb = q[2], xc = q[3];
      float y0 = n0[ctr * 32 + m];                          // L2-resident
      const float* yv = nv + (size_t)ctr * 96 + m * 3;
      float s0 = x0 * y0;
      float s1 = (xa * yv[0] + xb * yv[1] + xc * yv[2]) * 0.57735027f;
      Sn[row * 64 + m] = s0;
      Sn[row * 64 + 32 + m] = s1;
      ss0 += s0 * s0; ss1 += s1 * s1;
      float v00 = x0 * yv[0], v01 = x0 * yv[1], v02 = x0 * yv[2];
      float v10 = xa * y0,   v11 = xb * y0,   v12 = xc * y0;
      Vn[row * 192 + m * 3 + 0] = v00; Vn[row * 192 + m * 3 + 1] = v01; Vn[row * 192 + m * 3 + 2] = v02;
      Vn[row * 192 + 96 + m * 3 + 0] = v10; Vn[row * 192 + 96 + m * 3 + 1] = v11; Vn[row * 192 + 96 + m * 3 + 2] = v12;
      sv0 += v00 * v00 + v01 * v01 + v02 * v02;
      sv1 += v10 * v10 + v11 * v11 + v12 * v12;
    }
    ss0 += __shfl_xor(ss0, 16, 32);
    ss1 += __shfl_xor(ss1, 16, 32);
    sv0 += __shfl_xor(sv0, 16, 32);
    sv1 += __shfl_xor(sv1, 16, 32);
    float i0 = rsqrtf(ss0 / 32.f + 1e-5f);
    float i1 = rsqrtf(ss1 / 32.f + 1e-5f);
    float j0 = rsqrtf(sv0 / 96.f + 1e-5f);
    float j1 = rsqrtf(sv1 / 96.f + 1e-5f);
    for (int m = par; m < 32; m += 2) {
      float a = Sn[row * 64 + m] * i0;      Sn[row * 64 + m] = a;      F[row * 96 + m] = (_Float16)a;
      float b = Sn[row * 64 + 32 + m] * i1; Sn[row * 64 + 32 + m] = b; F[row * 96 + 32 + m] = (_Float16)b;
#pragma unroll
      for (int c = 0; c < 3; ++c) {
        Vn[row * 192 + m * 3 + c] *= j0;
        Vn[row * 192 + 96 + m * 3 + c] *= j1;
      }
    }
  }
  // conditioning block of the feature tile (streaming NT reads)
  for (int idx = l; idx < 16 * 8; idx += 32) {   // 16 rows x 8 f4 chunks
    int r = idx >> 3, c4 = (idx & 7) * 4;
    f4v v = ldnt4(cond + (e0 + r) * 32 + c4);
#pragma unroll
    for (int j = 0; j < 4; ++j) F[r * 96 + 64 + c4 + j] = (_Float16)v[j];
  }
  __syncthreads();

  // Wp1: 96 -> 256, silu -> H (f16)
  for (int ns = 0; ns < 16; ++ns) {
    v8f c = gemm_tile(F, 96, 96, pWp1, ns);
    int col = ns * 16 + (l & 15);
    float b = bp1[col];
#pragma unroll
    for (int r = 0; r < 8; ++r) {
      int row = r + ((l >> 4) << 3);
      H[row * 256 + col] = (_Float16)silu(c[r] + b);
    }
  }
  __syncthreads();

  float coeff = coeffp[0];
  float c_old = rsqrtf(coeff * coeff + 1.f);
  float c_new = coeff * c_old;

  // Wp2: 256 -> 192; scatter blended outputs straight from C layout.
  // Outputs are written once and never re-read -> NT stores keep them out of L2.
  for (int ns = 0; ns < 12; ++ns) {
    v8f acc = gemm_tile(H, 256, 256, pWp2, ns);
    int col = ns * 16 + (l & 15);
    float b = bp2[col];
#pragma unroll
    for (int r = 0; r < 8; ++r) {
      float val = acc[r] + b;
      int row = r + ((l >> 4) << 3);
      long e = e0 + row;
      float other = __shfl_xor(val, 1, 32);  // pair pw0<->pw1, pw2<->pw3 columns
      if (col < 128) {
        stnt(&scalar_out[e * 128 + col],
             c_old * ldnt(&scalar_latent[e * 128 + col]) + c_new * val);
      } else {
        int cc = col - 128;
        int m = cc >> 2, j = cc & 3;
        if (j == 0) {
          float eq0 = val * Sn[row * 64 + m] + other * Sn[row * 64 + 32 + m];
          stnt(&equiv_out[(e * 32 + m) * 4 + 0],
               c_old * ldnt(&equiv[(e * 32 + m) * 4 + 0]) + c_new * eq0);
        } else if (j == 2) {
#pragma unroll
          for (int cmp = 0; cmp < 3; ++cmp) {
            float ev = val * Vn[row * 192 + m * 3 + cmp] +
                       other * Vn[row * 192 + 96 + m * 3 + cmp];
            stnt(&equiv_out[(e * 32 + m) * 4 + 1 + cmp],
                 c_old * ldnt(&equiv[(e * 32 + m) * 4 + 1 + cmp]) + c_new * ev);
          }
        }
      }
    }
  }
}

// ---------------- host side ----------------
extern "C" void kernel_launch(void* const* d_in, const int* in_sizes, int n_in,
                              void* d_out, int out_size, void* d_ws, size_t ws_size,
                              hipStream_t stream) {
  const float* nodeinv = (const float*)d_in[0];
  const float* scalar  = (const float*)d_in[1];
  const float* equiv   = (const float*)d_in[2];
  const float* cond    = (const float*)d_in[3];
  const float* coeff   = (const float*)d_in[4];
  const float* Wenv1   = (const float*)d_in[5];
  const float* benv1   = (const float*)d_in[6];
  const float* Wenv2   = (const float*)d_in[7];
  const float* benv2   = (const float*)d_in[8];
  const float* Wq      = (const float*)d_in[9];
  const float* Wk      = (const float*)d_in[10];
  const float* Wp1     = (const float*)d_in[11];
  const float* bp1     = (const float*)d_in[12];
  const float* Wp2     = (const float*)d_in[13];
  const float* bp2     = (const float*)d_in[14];
  const int* edge_center = (const int*)d_in[15];
  // d_in[16] active_edges == arange(E): the residual update hits every edge.

  char* ws = (char*)d_ws;
  size_t off = 0;
  auto take = [&](size_t bytes) -> char* {
    char* p = ws + off;
    off = (off + bytes + 255) & ~(size_t)255;
    return p;
  };
  _Float16* pWenv1 = (_Float16*)take((size_t)192 * 128 * 2);
  _Float16* pWenv2 = (_Float16*)take((size_t)128 * 64 * 2);
  _Float16* pWk    = (_Float16*)take((size_t)128 * 512 * 2);
  _Float16* pWq    = (_Float16*)take((size_t)64 * 512 * 2);
  _Float16* pWp1   = (_Float16*)take((size_t)96 * 256 * 2);
  _Float16* pWp2   = (_Float16*)take((size_t)256 * 192 * 2);
  float* Qn    = (float*)take((size_t)N_NODES * 512 * 4);
  float* gates = (float*)take((size_t)E_EDGES * 64 * 4);
  float* wbuf  = (float*)take((size_t)E_EDGES * 32 * 4);
  float* denom = (float*)take((size_t)N_NODES * 32 * 4);
  float* n0acc = (float*)take((size_t)N_NODES * 32 * 4);
  float* nvacc = (float*)take((size_t)N_NODES * 96 * 4);

  auto pack = [&](const float* W, _Float16* P, int K, int Nout) {
    int total = K * Nout;
    pack_weights_k<<<(total + 255) / 256, 256, 0, stream>>>(W, P, K, Nout);
  };
  pack(Wenv1, pWenv1, 192, 128);
  pack(Wenv2, pWenv2, 128, 64);
  pack(Wk,    pWk,    128, 512);
  pack(Wq,    pWq,    64,  512);
  pack(Wp1,   pWp1,   96,  256);
  pack(Wp2,   pWp2,   256, 192);

  long nz = (long)N_NODES * 160;  // denom+n0acc+nvacc are contiguous allocations
  zero_k<<<(int)((nz + 255) / 256), 256, 0, stream>>>(denom, nz);

  qn_k<<<157, 128, 0, stream>>>(nodeinv, pWq, Qn);  // 625 row tiles

  phase1_k<<<E_EDGES / 16 / W1, 32 * W1, 0, stream>>>(
      scalar, equiv, edge_center, pWenv1, benv1, pWenv2, benv2, pWk, Qn,
      gates, wbuf, denom);

  long nsct = (long)E_EDGES * 32;
  scatter_k<<<(int)((nsct + 255) / 256), 256, 0, stream>>>(
      equiv, gates, wbuf, denom, edge_center, n0acc, nvacc);

  nodenorm_k<<<(N_NODES + 255) / 256, 256, 0, stream>>>(n0acc, nvacc);

  float* scalar_out = (float*)d_out;
  float* equiv_out  = (float*)d_out + (size_t)E_EDGES * 128;
  phase2_k<<<E_EDGES / 16 / W2, 32 * W2, 0, stream>>>(
      scalar, equiv, cond, edge_center, n0acc, nvacc, pWp1, bp1, pWp2, bp2,
      coeff, scalar_out, equiv_out);
}
